// Pointnet2MSG_57793079935581
// MI455X (gfx1250) — matrix operand order repaired and checked by reading, not compiled, not run on hardware
//
#include <hip/hip_runtime.h>

#define BATCH 4
#define NPC   16384

typedef __attribute__((ext_vector_type(16))) _Float16 v16h;
typedef __attribute__((ext_vector_type(8)))  float    v8f;

// ---------------------------------------------------------------------------
// Split pointcloud (B,N,6) -> xyz (B,N,3) [written to d_out] and feats0 (B,3,N)
// ---------------------------------------------------------------------------
__global__ void extract_kernel(const float* __restrict__ pc, float* __restrict__ xyz,
                               float* __restrict__ f0, int n, int total) {
  int i = blockIdx.x * blockDim.x + threadIdx.x;
  if (i >= total) return;               // total = B*n  (one thread per point)
  int b = i / n, p = i % n;
  const float* src = pc + (size_t)i * 6;
  float* dx = xyz + (size_t)i * 3;
  dx[0] = src[0]; dx[1] = src[1]; dx[2] = src[2];
  for (int c = 0; c < 3; ++c)
    f0[((size_t)b * 3 + c) * n + p] = src[3 + c];
}

// ---------------------------------------------------------------------------
// Farthest point sampling: one block per batch, sequential npoint rounds
// ---------------------------------------------------------------------------
__global__ __launch_bounds__(256)
void fps_kernel(const float* __restrict__ xyz, int n, int npoint,
                float* __restrict__ mind, int* __restrict__ idx) {
  int b = blockIdx.x, t = threadIdx.x;
  const float* X = xyz + (size_t)b * n * 3;
  float* md = mind + (size_t)b * n;
  __shared__ float sv[256];
  __shared__ int   si[256];
  for (int i = t; i < n; i += 256) md[i] = 1e10f;
  __syncthreads();
  int last = 0;
  for (int it = 0; it < npoint; ++it) {
    if (t == 0) idx[b * npoint + it] = last;
    float px = X[last * 3 + 0], py = X[last * 3 + 1], pz = X[last * 3 + 2];
    float bv = -1.0f; int bi = 0;
    for (int i = t; i < n; i += 256) {
      float dx = X[i * 3 + 0] - px, dy = X[i * 3 + 1] - py, dz = X[i * 3 + 2] - pz;
      float d = dx * dx + dy * dy + dz * dz;
      float m = md[i];
      if (d < m) { m = d; md[i] = m; }
      if (m > bv) { bv = m; bi = i; }
    }
    sv[t] = bv; si[t] = bi;
    __syncthreads();
    for (int s = 128; s > 0; s >>= 1) {
      if (t < s) {
        if (sv[t + s] > sv[t] || (sv[t + s] == sv[t] && si[t + s] < si[t])) {
          sv[t] = sv[t + s]; si[t] = si[t + s];
        }
      }
      __syncthreads();
    }
    last = si[0];
    __syncthreads();
  }
}

// gather sampled centers: out[b,i,:] = xyz[b, idx[b,i], :]
__global__ void gatherc_kernel(const float* __restrict__ xyz, const int* __restrict__ idx,
                               int n, int npoint, float* __restrict__ out, int total) {
  int i = blockIdx.x * blockDim.x + threadIdx.x;
  if (i >= total) return;               // total = B*npoint
  int b = i / npoint;
  int p = idx[i];
  const float* s = xyz + ((size_t)b * n + p) * 3;
  float* d = out + (size_t)i * 3;
  d[0] = s[0]; d[1] = s[1]; d[2] = s[2];
}

// ---------------------------------------------------------------------------
// Ball query: first ns ascending indices with d2 < r2, pad with first (or 0)
// ---------------------------------------------------------------------------
__global__ void ball_kernel(const float* __restrict__ sxyz, const float* __restrict__ cxyz,
                            int n, int npoint, float r2, int ns,
                            int* __restrict__ gidx, int total) {
  int g = blockIdx.x * blockDim.x + threadIdx.x;
  if (g >= total) return;               // total = B*npoint
  int b = g / npoint;
  const float* X = sxyz + (size_t)b * n * 3;
  float cx = cxyz[(size_t)g * 3 + 0], cy = cxyz[(size_t)g * 3 + 1], cz = cxyz[(size_t)g * 3 + 2];
  int* out = gidx + (size_t)g * ns;
  int cnt = 0, first = -1;
  for (int i = 0; i < n && cnt < ns; ++i) {
    float dx = X[i * 3 + 0] - cx, dy = X[i * 3 + 1] - cy, dz = X[i * 3 + 2] - cz;
    if (dx * dx + dy * dy + dz * dz < r2) {
      if (cnt == 0) first = i;
      out[cnt++] = i;
    }
  }
  int pad = (first >= 0) ? first : 0;
  for (; cnt < ns; ++cnt) out[cnt] = pad;
}

// build grouped tensor g (B, 3+Cf, npoint, ns): relative xyz then gathered feats
__global__ void group_kernel(const float* __restrict__ sxyz, const float* __restrict__ cxyz,
                             const float* __restrict__ feats, const int* __restrict__ gidx,
                             int n, int npoint, int ns, int Cf,
                             float* __restrict__ g, int total) {
  int i = blockIdx.x * blockDim.x + threadIdx.x;
  if (i >= total) return;               // total = B*(3+Cf)*npoint*ns
  int C = 3 + Cf;
  int s = i % ns;
  int m = (i / ns) % npoint;
  int c = (i / (ns * npoint)) % C;
  int b = i / (ns * npoint * C);
  int p = gidx[((size_t)(b * npoint + m)) * ns + s];
  float v;
  if (c < 3)
    v = sxyz[((size_t)b * n + p) * 3 + c] - cxyz[((size_t)b * npoint + m) * 3 + c];
  else
    v = feats[((size_t)b * Cf + (c - 3)) * n + p];
  g[i] = v;
}

// ---------------------------------------------------------------------------
// Operand pre-swizzle into a per-lane WMMA operand image (f16, zero-padded; same k order in A and B).
// A layout per k-tile/o-tile: lanes 0-15 hold K 0..15, lanes 16-31 K 16..31, M = lane&15.
// B layout per k-tile/c-tile: lanes 0-15 hold K 0..15, lanes 16-31 K 16..31.
// Lane-contiguous: ((tile)*32 + lane)*16 + e  ->  32B per lane, b128 loads.
// ---------------------------------------------------------------------------
__global__ void packA_kernel(const float* __restrict__ W, int K, int O, int Ot,
                             _Float16* __restrict__ pA, int total) {
  int i = blockIdx.x * blockDim.x + threadIdx.x;
  if (i >= total) return;               // total = Ktiles*Ot*512
  int e = i & 15, lane = (i >> 4) & 31, tile = i >> 9;
  int ot = tile % Ot, kt = tile / Ot;
  int grp = lane >> 4, idx = lane & 15;
  int ka = kt * 32 + grp * 16 + e;
  int o = ot * 16 + idx;
  pA[i] = (ka < K && o < O) ? (_Float16)W[(size_t)ka * O + o] : (_Float16)0.0f;
}

__global__ void packB_kernel(const float* __restrict__ X, int K, int MS,
                             int Ktiles, int Ct, _Float16* __restrict__ pB, int total) {
  int i = blockIdx.x * blockDim.x + threadIdx.x;
  if (i >= total) return;               // total = B*Ktiles*Ct*512
  int e = i & 15, lane = (i >> 4) & 31;
  int r = i >> 9;
  int ct = r % Ct; r /= Ct;
  int kt = r % Ktiles;
  int b = r / Ktiles;
  int k = kt * 32 + (lane >> 4) * 16 + e;
  int col = ct * 16 + (lane & 15);
  pB[i] = (k < K) ? (_Float16)X[((size_t)b * K + k) * MS + col] : (_Float16)0.0f;
}

// ---------------------------------------------------------------------------
// WMMA GEMM on pre-swizzled operands: Y[b,o,col] = sum_k W[k,o]*X[b,k,col].
// 4 waves/block, each wave -> 16(O) x 32(cols) via two accumulators.
// MS is always a multiple of 128 here -> no column guards, EXEC all-1s.
// ---------------------------------------------------------------------------
__global__ __launch_bounds__(128)
void gemm_wmma(const _Float16* __restrict__ pA, const _Float16* __restrict__ pB,
               float* __restrict__ Y, int Ktiles, int Ot, int O, int MS) {
  const int lane = threadIdx.x & 31;
  const int wave = threadIdx.x >> 5;
  const int grp  = lane >> 4, idx = lane & 15;
  const int b    = blockIdx.z;
  const int ot   = blockIdx.y;
  const int ct0  = (blockIdx.x * 4 + wave) * 2;   // two 16-col tiles per wave
  const int Ct   = MS >> 4;

  const _Float16* Ap = pA + (size_t)ot * 512 + lane * 16;
  const _Float16* Bp = pB + (((size_t)b * Ktiles * Ct + ct0) * 32 + lane) * 16;
  const size_t strideA = (size_t)Ot * 512;
  const size_t strideB = (size_t)Ct * 512;

  v8f acc0 = {}, acc1 = {};
  for (int kt = 0; kt < Ktiles; ++kt) {
    v16h a  = *(const v16h*)(Ap + (size_t)kt * strideA);
    v16h b0 = *(const v16h*)(Bp + (size_t)kt * strideB);
    v16h b1 = *(const v16h*)(Bp + (size_t)kt * strideB + 512);
    acc0 = __builtin_amdgcn_wmma_f32_16x16x32_f16(false, a, false, b0,
                                                  (short)0, acc0, false, false);
    acc1 = __builtin_amdgcn_wmma_f32_16x16x32_f16(false, a, false, b1,
                                                  (short)0, acc1, false, false);
  }

  const int col = ct0 * 16 + idx;
#pragma unroll
  for (int r = 0; r < 8; ++r) {
    int o = ot * 16 + r + grp * 8;      // D layout: vgpr r -> row r (+8 for hi lanes)
    if (o < O) {
      float* yp = Y + ((size_t)b * O + o) * MS + col;
      yp[0]  = acc0[r];
      yp[16] = acc1[r];
    }
  }
}

// ---------------------------------------------------------------------------
// BatchNorm (training-mode batch stats) + ReLU
// ---------------------------------------------------------------------------
__global__ __launch_bounds__(256)
void bnstats_kernel(const float* __restrict__ Y, int O, int MS, int Bn,
                    float* __restrict__ mean, float* __restrict__ rstd) {
  int o = blockIdx.x, t = threadIdx.x;
  __shared__ float ss[256], sq[256];
  float s = 0.f, q = 0.f;
  for (int b = 0; b < Bn; ++b) {
    const float* p = Y + ((size_t)b * O + o) * MS;
    for (int i = t; i < MS; i += 256) { float v = p[i]; s += v; q += v * v; }
  }
  ss[t] = s; sq[t] = q;
  __syncthreads();
  for (int st = 128; st > 0; st >>= 1) {
    if (t < st) { ss[t] += ss[t + st]; sq[t] += sq[t + st]; }
    __syncthreads();
  }
  if (t == 0) {
    float cnt = (float)Bn * (float)MS;
    float m = ss[0] / cnt;
    float var = sq[0] / cnt - m * m;
    mean[o] = m;
    rstd[o] = rsqrtf(var + 1e-3f);
  }
}

__global__ void bnapply_kernel(const float* __restrict__ Y, float* __restrict__ dst,
                               const float* __restrict__ gamma, const float* __restrict__ beta,
                               const float* __restrict__ mean, const float* __restrict__ rstd,
                               int O, int MS, int total) {
  int i = blockIdx.x * blockDim.x + threadIdx.x;
  if (i >= total) return;
  int o = (i / MS) % O;
  float v = gamma[o] * (Y[i] - mean[o]) * rstd[o] + beta[o];
  dst[i] = v > 0.f ? v : 0.f;
}

// max over samples: (B,O,M,S) -> write into feats buffer at channel offset
__global__ void maxpool_kernel(const float* __restrict__ X, float* __restrict__ dst,
                               int O, int M, int S, int dstC, int coff, int total) {
  int i = blockIdx.x * blockDim.x + threadIdx.x;
  if (i >= total) return;               // total = B*O*M
  int m = i % M;
  int o = (i / M) % O;
  int b = i / (M * O);
  const float* p = X + (((size_t)b * O + o) * M + m) * S;
  float mx = p[0];
  for (int s = 1; s < S; ++s) mx = fmaxf(mx, p[s]);
  dst[((size_t)b * dstC + coff + o) * M + m] = mx;
}

// ---------------------------------------------------------------------------
// 3-NN (squared distances) + inverse-distance weights
// ---------------------------------------------------------------------------
__global__ void threenn_kernel(const float* __restrict__ q, const float* __restrict__ s,
                               int nq, int nsrc, int* __restrict__ nnidx,
                               float* __restrict__ nnw, int total) {
  int g = blockIdx.x * blockDim.x + threadIdx.x;
  if (g >= total) return;               // total = B*nq
  int b = g / nq;
  float qx = q[(size_t)g * 3 + 0], qy = q[(size_t)g * 3 + 1], qz = q[(size_t)g * 3 + 2];
  const float* S = s + (size_t)b * nsrc * 3;
  float d0 = 3e38f, d1 = 3e38f, d2 = 3e38f;
  int i0 = 0, i1 = 0, i2 = 0;
  for (int j = 0; j < nsrc; ++j) {
    float dx = S[j * 3 + 0] - qx, dy = S[j * 3 + 1] - qy, dz = S[j * 3 + 2] - qz;
    float d = dx * dx + dy * dy + dz * dz;
    if (d < d0)      { d2 = d1; i2 = i1; d1 = d0; i1 = i0; d0 = d; i0 = j; }
    else if (d < d1) { d2 = d1; i2 = i1; d1 = d;  i1 = j; }
    else if (d < d2) { d2 = d;  i2 = j; }
  }
  float r0 = 1.f / (d0 + 1e-8f), r1 = 1.f / (d1 + 1e-8f), r2 = 1.f / (d2 + 1e-8f);
  float inv = 1.f / (r0 + r1 + r2);
  nnidx[(size_t)g * 3 + 0] = i0; nnidx[(size_t)g * 3 + 1] = i1; nnidx[(size_t)g * 3 + 2] = i2;
  nnw[(size_t)g * 3 + 0] = r0 * inv; nnw[(size_t)g * 3 + 1] = r1 * inv; nnw[(size_t)g * 3 + 2] = r2 * inv;
}

// interpolate: dst[b, c, i] (stride dstC) = sum_k f[b,c,idx_k] * w_k
__global__ void interp_kernel(const float* __restrict__ f, const int* __restrict__ nnidx,
                              const float* __restrict__ nnw, int C, int nsrc, int nq,
                              int dstC, float* __restrict__ dst, int total) {
  int i = blockIdx.x * blockDim.x + threadIdx.x;
  if (i >= total) return;               // total = B*C*nq
  int p = i % nq;
  int c = (i / nq) % C;
  int b = i / (nq * C);
  const int*   id = nnidx + ((size_t)b * nq + p) * 3;
  const float* w  = nnw   + ((size_t)b * nq + p) * 3;
  const float* fc = f + ((size_t)b * C + c) * nsrc;
  float v = fc[id[0]] * w[0] + fc[id[1]] * w[1] + fc[id[2]] * w[2];
  dst[((size_t)b * dstC + c) * nq + p] = v;
}

// copy channel block: dst[b, coff+c, m] (stride dstC) = src[b, c, m]
__global__ void copych_kernel(const float* __restrict__ src, float* __restrict__ dst,
                              int C, int M, int dstC, int coff, int total) {
  int i = blockIdx.x * blockDim.x + threadIdx.x;
  if (i >= total) return;               // total = B*C*M
  int m = i % M;
  int c = (i / M) % C;
  int b = i / (M * C);
  dst[((size_t)b * dstC + coff + c) * M + m] = src[i];
}

// ---------------------------------------------------------------------------
// Host orchestration
// ---------------------------------------------------------------------------
struct Lyr { const float *W, *g, *b; int ci, co; };

extern "C" void kernel_launch(void* const* d_in, const int* in_sizes, int n_in,
                              void* d_out, int out_size, void* d_ws, size_t ws_size,
                              hipStream_t stream) {
  (void)in_sizes; (void)n_in; (void)out_size; (void)ws_size;
  const int B = BATCH, N = NPC;
  const float* pc = (const float*)d_in[0];
  float* out_xyz  = (float*)d_out;                       // (B,N,3)
  float* out_feat = (float*)d_out + (size_t)B * N * 3;   // (B,128,N)

  // deterministic bump allocator over workspace
  char* base = (char*)d_ws;
  size_t off = 0;
  auto alloc = [&](size_t bytes) -> void* {
    void* p = base + off;
    off = (off + bytes + 255) & ~(size_t)255;
    return p;
  };
  float* xyz1    = (float*)alloc((size_t)B * 1024 * 3 * 4);
  float* xyz2    = (float*)alloc((size_t)B * 256 * 3 * 4);
  float* xyz3    = (float*)alloc((size_t)B * 64 * 3 * 4);
  float* feats0  = (float*)alloc((size_t)B * 3 * N * 4);
  float* feats1  = (float*)alloc((size_t)B * 96 * 1024 * 4);
  float* feats2  = (float*)alloc((size_t)B * 256 * 256 * 4);
  float* feats3  = (float*)alloc((size_t)B * 512 * 64 * 4);
  float* feats2b = (float*)alloc((size_t)B * 512 * 256 * 4);
  float* feats1b = (float*)alloc((size_t)B * 256 * 1024 * 4);
  float* fpsmin  = (float*)alloc((size_t)B * N * 4);
  float* meanb   = (float*)alloc(512 * 4);
  float* rstdb   = (float*)alloc(512 * 4);
  int*   fpsidx  = (int*)alloc((size_t)B * 1024 * 4);
  int*   gidx    = (int*)alloc((size_t)B * 1024 * 32 * 4);
  int*   nnidx   = (int*)alloc((size_t)B * N * 3 * 4);
  float* nnw     = (float*)alloc((size_t)B * N * 3 * 4);
  float* bufA    = (float*)alloc((size_t)B * 259 * N * 4);   // biggest concat/interp
  float* bufB    = (float*)alloc((size_t)B * 128 * N * 4);
  float* bufC    = (float*)alloc((size_t)B * 128 * N * 4);
  _Float16* packA = (_Float16*)alloc((size_t)768 * 512 * 2);       // Kpad*Opad max
  _Float16* packB = (_Float16*)alloc((size_t)B * 288 * N * 2);     // B*Kpad*MS max

  // ---- parameter table (d_in order: pointcloud, sa stages/scales/layers, fp) ----
  static const int sas[3][2][4] = {
    {{6, 16, 16, 32}, {6, 32, 32, 64}},
    {{99, 64, 64, 128}, {99, 64, 96, 128}},
    {{259, 128, 196, 256}, {259, 128, 196, 256}}};
  static const int fps_[3][3] = {{259, 128, 128}, {608, 256, 256}, {768, 512, 512}};
  Lyr sa[3][2][3], fp[3][2];
  int pi = 1;
  for (int si = 0; si < 3; ++si)
    for (int j = 0; j < 2; ++j)
      for (int l = 0; l < 3; ++l) {
        sa[si][j][l] = {(const float*)d_in[pi], (const float*)d_in[pi + 1],
                        (const float*)d_in[pi + 2], sas[si][j][l], sas[si][j][l + 1]};
        pi += 3;
      }
  for (int i = 0; i < 3; ++i)
    for (int l = 0; l < 2; ++l) {
      fp[i][l] = {(const float*)d_in[pi], (const float*)d_in[pi + 1],
                  (const float*)d_in[pi + 2], fps_[i][l], fps_[i][l + 1]};
      pi += 3;
    }

  auto run_mlp = [&](const Lyr* L, int nl, const float* in0, float* finalDst,
                     int MS) -> const float* {
    const float* cur = in0;
    float* bufs[2] = {bufB, bufC};
    for (int l = 0; l < nl; ++l) {
      float* o = bufs[l & 1];
      int K = L[l].ci, O = L[l].co;
      int Ktiles = (K + 31) / 32, Ot = (O + 15) / 16, Ct = MS / 16;
      int totA = Ktiles * Ot * 512;
      packA_kernel<<<(totA + 255) / 256, 256, 0, stream>>>(L[l].W, K, O, Ot, packA, totA);
      int totB = B * Ktiles * Ct * 512;
      packB_kernel<<<(totB + 255) / 256, 256, 0, stream>>>(cur, K, MS, Ktiles, Ct,
                                                           packB, totB);
      gemm_wmma<<<dim3(MS / 128, Ot, B), 128, 0, stream>>>(packA, packB, o,
                                                           Ktiles, Ot, O, MS);
      bnstats_kernel<<<O, 256, 0, stream>>>(o, O, MS, B, meanb, rstdb);
      float* dst = (l == nl - 1 && finalDst) ? finalDst : o;
      int tot = B * O * MS;
      bnapply_kernel<<<(tot + 255) / 256, 256, 0, stream>>>(o, dst, L[l].g, L[l].b,
                                                            meanb, rstdb, O, MS, tot);
      cur = dst;
    }
    return cur;
  };

  // ---- split input ----
  {
    int tot = B * N;
    extract_kernel<<<(tot + 255) / 256, 256, 0, stream>>>(pc, out_xyz, feats0, N, tot);
  }

  const float* lx[4] = {out_xyz, xyz1, xyz2, xyz3};
  float*       lxm[4] = {out_xyz, xyz1, xyz2, xyz3};
  const float* lf[4] = {feats0, feats1, feats2, feats3};
  float*       lfm[4] = {feats0, feats1, feats2, feats3};
  const int ln[4] = {N, 1024, 256, 64};
  const int lc[4] = {3, 96, 256, 512};
  static const float R2[3][2] = {{0.04f, 0.16f}, {0.16f, 0.64f}, {0.64f, 2.56f}};
  static const int   NSAMP[3][2] = {{16, 32}, {16, 32}, {16, 32}};

  // ---- set abstraction stages ----
  for (int si = 0; si < 3; ++si) {
    int n = ln[si], np = ln[si + 1];
    fps_kernel<<<B, 256, 0, stream>>>(lx[si], n, np, fpsmin, fpsidx);
    int totc = B * np;
    gatherc_kernel<<<(totc + 255) / 256, 256, 0, stream>>>(lx[si], fpsidx, n, np,
                                                           lxm[si + 1], totc);
    for (int j = 0; j < 2; ++j) {
      int ns = NSAMP[si][j];
      ball_kernel<<<(totc + 255) / 256, 256, 0, stream>>>(lx[si], lx[si + 1], n, np,
                                                          R2[si][j], ns, gidx, totc);
      int Cin = 3 + lc[si];
      int totg = B * Cin * np * ns;
      group_kernel<<<(totg + 255) / 256, 256, 0, stream>>>(lx[si], lx[si + 1], lf[si],
                                                           gidx, n, np, ns, lc[si],
                                                           bufA, totg);
      const float* mo = run_mlp(sa[si][j], 3, bufA, nullptr, np * ns);
      int O = sa[si][j][2].co;
      int coff = (j == 0) ? 0 : sa[si][0][2].co;
      int totm = B * O * np;
      maxpool_kernel<<<(totm + 255) / 256, 256, 0, stream>>>(mo, lfm[si + 1], O, np, ns,
                                                             lc[si + 1], coff, totm);
    }
  }

  // ---- feature propagation (i = 2, 1, 0) ----
  {  // i = 2: 256 queries from 64 sources; 512+256 -> fp[2] -> feats2b (512 ch)
    int nq = 256, nsrc = 64;
    threenn_kernel<<<(B * nq + 255) / 256, 256, 0, stream>>>(xyz2, xyz3, nq, nsrc,
                                                             nnidx, nnw, B * nq);
    int ti = B * 512 * nq;
    interp_kernel<<<(ti + 255) / 256, 256, 0, stream>>>(feats3, nnidx, nnw, 512, nsrc,
                                                        nq, 768, bufA, ti);
    int tc = B * 256 * nq;
    copych_kernel<<<(tc + 255) / 256, 256, 0, stream>>>(feats2, bufA, 256, nq, 768, 512, tc);
    run_mlp(fp[2], 2, bufA, feats2b, nq);
  }
  {  // i = 1: 1024 queries from 256 sources; 512+96 -> fp[1] -> feats1b (256 ch)
    int nq = 1024, nsrc = 256;
    threenn_kernel<<<(B * nq + 255) / 256, 256, 0, stream>>>(xyz1, xyz2, nq, nsrc,
                                                             nnidx, nnw, B * nq);
    int ti = B * 512 * nq;
    interp_kernel<<<(ti + 255) / 256, 256, 0, stream>>>(feats2b, nnidx, nnw, 512, nsrc,
                                                        nq, 608, bufA, ti);
    int tc = B * 96 * nq;
    copych_kernel<<<(tc + 255) / 256, 256, 0, stream>>>(feats1, bufA, 96, nq, 608, 512, tc);
    run_mlp(fp[1], 2, bufA, feats1b, nq);
  }
  {  // i = 0: N queries from 1024 sources; 256+3 -> fp[0] -> d_out feats (128 ch)
    int nq = N, nsrc = 1024;
    threenn_kernel<<<(B * nq + 255) / 256, 256, 0, stream>>>(out_xyz, xyz1, nq, nsrc,
                                                             nnidx, nnw, B * nq);
    int ti = B * 256 * nq;
    interp_kernel<<<(ti + 255) / 256, 256, 0, stream>>>(feats1b, nnidx, nnw, 256, nsrc,
                                                        nq, 259, bufA, ti);
    int tc = B * 3 * nq;
    copych_kernel<<<(tc + 255) / 256, 256, 0, stream>>>(feats0, bufA, 3, nq, 259, 256, tc);
    run_mlp(fp[0], 2, bufA, out_feat, nq);
  }
}
